// DFT_46986942218954
// MI455X (gfx1250) — matrix operand order repaired and checked by reading, not compiled, not run on hardware
//
#include <hip/hip_runtime.h>

#define NDIM 2048   // DFT size (K and output columns)
#define BDIM 8192   // batch rows
#define BM 128
#define BN 128
#define BK 32
#define LDK 40      // LDS row pitch in halves (80B: 16B-aligned, bank-spread)
#define KTILES (NDIM / BK)

typedef _Float16 v16h __attribute__((ext_vector_type(16)));
typedef _Float16 v8h  __attribute__((ext_vector_type(8)));
typedef _Float16 h4   __attribute__((ext_vector_type(4)));
typedef float    v8f  __attribute__((ext_vector_type(8)));

// Global f32 -> LDS f16 staging: 128 rows x 32 cols, 256 threads, float4 loads.
__device__ __forceinline__ void stage_tile(const float* __restrict__ g, int row0, int kb,
                                           _Float16* __restrict__ s, int t) {
#pragma unroll
  for (int j = 0; j < 4; ++j) {
    int idx = t + 256 * j;        // 0..1023 -> 128x8 float4 tile
    int r   = idx >> 3;           // row in tile
    int c   = (idx & 7) << 2;     // col (floats)
    const float4 v = *(const float4*)(g + (size_t)(row0 + r) * NDIM + kb + c);
    h4 hv = { (_Float16)v.x, (_Float16)v.y, (_Float16)v.z, (_Float16)v.w };
    *(h4*)(s + r * LDK + c) = hv;
  }
}

// Same, but writes both +x and -x tiles from one global load; the negation
// folds into the v_cvt_f16_f32 source modifier (free in VALU).
__device__ __forceinline__ void stage_tile_pm(const float* __restrict__ g, int row0, int kb,
                                              _Float16* __restrict__ sp,
                                              _Float16* __restrict__ sn, int t) {
#pragma unroll
  for (int j = 0; j < 4; ++j) {
    int idx = t + 256 * j;
    int r   = idx >> 3;
    int c   = (idx & 7) << 2;
    const float4 v = *(const float4*)(g + (size_t)(row0 + r) * NDIM + kb + c);
    h4 hp = { (_Float16)v.x,  (_Float16)v.y,  (_Float16)v.z,  (_Float16)v.w };
    h4 hn = { (_Float16)-v.x, (_Float16)-v.y, (_Float16)-v.z, (_Float16)-v.w };
    *(h4*)(sp + r * LDK + c) = hp;
    *(h4*)(sn + r * LDK + c) = hn;
  }
}

// A fragment (16x32 f16, MxK): lane m holds row m; VGPR0-3 = K base..base+7,
// VGPR4-7 = K base+16..base+23, base = (lane>=16)*8.  (ISA 7.12.2)
__device__ __forceinline__ v16h frag_A(const _Float16* __restrict__ s, int lane, int m0) {
  int m  = m0 + (lane & 15);
  int hi = (lane >> 4) << 3;                         // 0 or 8
  v8h lo = *(const v8h*)(s + m * LDK + hi);          // K hi..hi+7
  v8h hh = *(const v8h*)(s + m * LDK + 16 + hi);     // K hi+16..hi+23
  return __builtin_shufflevector(lo, hh, 0, 1, 2, 3, 4, 5, 6, 7,
                                 8, 9, 10, 11, 12, 13, 14, 15);
}

// B fragment (32x16 f16, KxN): lane n holds column n; elements 0-7 = K base..base+7,
// elements 8-15 = K base+16..base+23, base = (lane>=16)*8 (matches frag_A). W is symmetric, so
// column n of W == row n of W; s is staged [n][k].
__device__ __forceinline__ v16h frag_B(const _Float16* __restrict__ s, int lane, int n0) {
  int n  = n0 + (lane & 15);
  int hi = (lane >> 4) << 3;                         // 0 or 8
  v8h lo = *(const v8h*)(s + n * LDK + hi);
  v8h hh = *(const v8h*)(s + n * LDK + 16 + hi);
  return __builtin_shufflevector(lo, hh, 0, 1, 2, 3, 4, 5, 6, 7,
                                 8, 9, 10, 11, 12, 13, 14, 15);
}

__device__ __forceinline__ v8f wmma(v16h a, v16h b, v8f c) {
  return __builtin_amdgcn_wmma_f32_16x16x32_f16(false, a, false, b,
                                                (short)0, c, false, false);
}

__global__ __launch_bounds__(256)
void dft_wmma_f16_kernel(const float* __restrict__ xr, const float* __restrict__ xi,
                         const float* __restrict__ wr, const float* __restrict__ wi,
                         float* __restrict__ out) {
  __shared__ _Float16 sXr [2][BM * LDK];
  __shared__ _Float16 sXi [2][BM * LDK];
  __shared__ _Float16 sXin[2][BM * LDK];   // -x_i, negation folded into cvt
  __shared__ _Float16 sWr [2][BN * LDK];
  __shared__ _Float16 sWi [2][BN * LDK];

  const int t    = threadIdx.x;
  const int lane = t & 31;
  const int wave = t >> 5;
  const int wm   = (wave & 3) * 32;   // 4 wave rows of 32
  const int wn   = (wave >> 2) * 64;  // 2 wave cols of 64
  const int m0   = blockIdx.y * BM;
  const int n0   = blockIdx.x * BN;

  stage_tile   (xr, m0, 0, sXr[0], t);
  stage_tile_pm(xi, m0, 0, sXi[0], sXin[0], t);
  stage_tile   (wr, n0, 0, sWr[0], t);  // symmetric W: rows over k == cols over k
  stage_tile   (wi, n0, 0, sWi[0], t);
  __syncthreads();

  const v8f zero = {0.f, 0.f, 0.f, 0.f, 0.f, 0.f, 0.f, 0.f};
  v8f zr[2][4], zi[2][4];
#pragma unroll
  for (int mi = 0; mi < 2; ++mi)
#pragma unroll
    for (int ni = 0; ni < 4; ++ni) { zr[mi][ni] = zero; zi[mi][ni] = zero; }

  for (int kt = 0; kt < KTILES; ++kt) {
    const int cur = kt & 1;
    if (kt + 1 < KTILES) {            // prefetch next stage while computing
      const int kb = (kt + 1) * BK;
      stage_tile   (xr, m0, kb, sXr[cur ^ 1], t);
      stage_tile_pm(xi, m0, kb, sXi[cur ^ 1], sXin[cur ^ 1], t);
      stage_tile   (wr, n0, kb, sWr[cur ^ 1], t);
      stage_tile   (wi, n0, kb, sWi[cur ^ 1], t);
    }

    const _Float16* bXr  = sXr [cur];
    const _Float16* bXi  = sXi [cur];
    const _Float16* bXin = sXin[cur];
    const _Float16* bWr  = sWr [cur];
    const _Float16* bWi  = sWi [cur];

    v16h ar[2], ai[2], nai[2];
#pragma unroll
    for (int mi = 0; mi < 2; ++mi) {
      ar[mi]  = frag_A(bXr,  lane, wm + 16 * mi);
      ai[mi]  = frag_A(bXi,  lane, wm + 16 * mi);
      nai[mi] = frag_A(bXin, lane, wm + 16 * mi);  // pre-negated in LDS
    }

#pragma unroll
    for (int ni = 0; ni < 4; ++ni) {
      v16h br = frag_B(bWr, lane, wn + 16 * ni);
      v16h bi = frag_B(bWi, lane, wn + 16 * ni);
#pragma unroll
      for (int mi = 0; mi < 2; ++mi) {
        zr[mi][ni] = wmma(ar[mi],  br, zr[mi][ni]);   // + x_r*W_r
        zr[mi][ni] = wmma(nai[mi], bi, zr[mi][ni]);   // - x_i*W_i
        zi[mi][ni] = wmma(ai[mi],  br, zi[mi][ni]);   // + x_i*W_r
        zi[mi][ni] = wmma(ar[mi],  bi, zi[mi][ni]);   // + x_r*W_i
      }
    }
    __syncthreads();
  }

  // C/D layout: VGPR v -> row v + 8*(lane>=16); col = lane&15  (ISA 7.12.2)
  float* outR = out;
  float* outI = out + (size_t)BDIM * NDIM;
  const int colBase = n0 + wn + (lane & 15);
  const int rowBase = m0 + wm + ((lane >> 4) << 3);
#pragma unroll
  for (int mi = 0; mi < 2; ++mi) {
#pragma unroll
    for (int ni = 0; ni < 4; ++ni) {
      const int col = colBase + 16 * ni;
#pragma unroll
      for (int v = 0; v < 8; ++v) {
        const size_t row = (size_t)(rowBase + 16 * mi + v);
        outR[row * NDIM + col] = zr[mi][ni][v];
        outI[row * NDIM + col] = zi[mi][ni][v];
      }
    }
  }
}

extern "C" void kernel_launch(void* const* d_in, const int* in_sizes, int n_in,
                              void* d_out, int out_size, void* d_ws, size_t ws_size,
                              hipStream_t stream) {
  const float* xr = (const float*)d_in[0];
  const float* xi = (const float*)d_in[1];
  const float* wr = (const float*)d_in[2];
  const float* wi = (const float*)d_in[3];
  dim3 grid(NDIM / BN, BDIM / BM);   // 16 x 64 workgroups
  dft_wmma_f16_kernel<<<grid, dim3(256), 0, stream>>>(xr, xi, wr, wi, (float*)d_out);
}
